// GraphSAGEModel_61701500174760
// MI455X (gfx1250) — compile-verified, matrix operand order checked by reading, not run on hardware
//
#include <hip/hip_runtime.h>
#include <hip/hip_bf16.h>

#define N_NODES 100000
#define IN_DIM  128
#define HID     256
#define N_EDGES 1600000
#define BN_EPS  1e-5f

typedef __attribute__((ext_vector_type(16))) _Float16 v16h;
typedef __attribute__((ext_vector_type(8)))  float    v8f;

union Frag16 { uint4 u[2]; v16h h; };

// ---------------------------------------------------------------- utilities
__global__ void zero_f32(float* __restrict__ p, int n) {
    int i = blockIdx.x * blockDim.x + threadIdx.x;
    if (i < n) p[i] = 0.0f;
}

__global__ void deg_kernel(const int* __restrict__ dst, float* __restrict__ deg, int nE) {
    int e = blockIdx.x * blockDim.x + threadIdx.x;
    if (e < nE) atomicAdd(&deg[dst[e]], 1.0f);
}

// one thread per (edge, 4-float chunk): agg[dst] += x[src]
__global__ void scatter_add(const float* __restrict__ x, const int* __restrict__ src,
                            const int* __restrict__ dst, float* __restrict__ agg,
                            int D4, int total) {
    int gid = blockIdx.x * blockDim.x + threadIdx.x;
    if (gid >= total) return;
    int e = gid / D4;
    int c = gid - e * D4;
    int s = src[e], d = dst[e];
    float4 v = ((const float4*)x)[(size_t)s * D4 + c];
    float* a = agg + ((size_t)d * D4 + c) * 4;
    atomicAdd(a + 0, v.x);
    atomicAdd(a + 1, v.y);
    atomicAdd(a + 2, v.z);
    atomicAdd(a + 3, v.w);
}

__global__ void norm_agg(float* __restrict__ agg, const float* __restrict__ deg,
                         int D4, int total) {
    int gid = blockIdx.x * blockDim.x + threadIdx.x;
    if (gid >= total) return;
    int node = gid / D4;
    float inv = 1.0f / fmaxf(deg[node], 1.0f);
    float4* a = (float4*)agg + gid;
    float4 v = *a;
    v.x *= inv; v.y *= inv; v.z *= inv; v.w *= inv;
    *a = v;
}

// ---------------------------------------------------------------- WMMA GEMM
// out[M,NCOLS] = [A0 | A1] @ [W0 ; W1] + bias  (optional ReLU)
// A0:[M,D0] f32, A1:[M,D1] f32 (nullable), W0:[D0,NCOLS], W1:[D1,NCOLS].
// Block = 128 threads = 4 waves; block covers 64 rows x NCOLS cols.
template <int NCOLS>
__global__ __launch_bounds__(128) void sage_gemm(
    const float* __restrict__ A0, const float* __restrict__ A1,
    const float* __restrict__ W0, const float* __restrict__ W1,
    const float* __restrict__ bias, float* __restrict__ out,
    int nRows, int D0, int D1, int doRelu)
{
    constexpr int NT = NCOLS / 16;
    __shared__ __align__(16) _Float16 lA[64 * 32];       // [row][k]
    __shared__ __align__(16) _Float16 lB[NCOLS * 32];    // [col][k] (W transposed)

    const int tid  = threadIdx.x;
    const int wave = tid >> 5;
    const int lane = tid & 31;
    const int rowBase = blockIdx.x * 64;

    v8f acc[NT] = {};

    const int K = D0 + D1;
    const int ksteps = K >> 5;

    for (int ks = 0; ks < ksteps; ++ks) {
        const int  kg     = ks << 5;
        const bool second = (kg >= D0);
        const float* Asrc = second ? A1 : A0;
        const float* Wsrc = second ? W1 : W0;
        const int  Dsrc   = second ? D1 : D0;
        const int  kOff   = second ? (kg - D0) : kg;

        __syncthreads();
        // stage A tile: 64 rows x 32 k, f32 -> f16 (coalesced along k)
        #pragma unroll
        for (int i = 0; i < 16; ++i) {
            int idx = i * 128 + tid;
            int r = idx >> 5, k = idx & 31;
            int g = rowBase + r;
            float v = (g < nRows) ? Asrc[(size_t)g * Dsrc + kOff + k] : 0.0f;
            lA[r * 32 + k] = (_Float16)v;
        }
        // stage B tile: W[k][n] -> lB[n][k] (coalesced along n)
        #pragma unroll
        for (int i = 0; i < NCOLS / 4; ++i) {
            int idx = i * 128 + tid;
            int k = idx / NCOLS, n = idx % NCOLS;
            lB[n * 32 + k] = (_Float16)Wsrc[(size_t)(kOff + k) * NCOLS + n];
        }
        // prefetch next K-slab of A while we compute (global_prefetch_b8)
        if (ks + 1 < ksteps) {
            const bool s2 = (kg + 32) >= D0;
            const float* A2 = s2 ? A1 : A0;
            const int Ds2 = s2 ? D1 : D0;
            const int kO2 = s2 ? (kg + 32 - D0) : (kg + 32);
            int g = rowBase + (tid >> 1);
            if (g < nRows)
                __builtin_prefetch(&A2[(size_t)g * Ds2 + kO2 + (tid & 1) * 16], 0, 1);
        }
        __syncthreads();

        // A fragment (ISA 16-bit 16x32 layout): lane half selects K group 0/8
        const int half = lane >> 4;
        const int fRow = lane & 15;
        const int kb   = half * 8;            // f16 offset: VGPR0-3 -> K kb..kb+7
        Frag16 fa;
        const _Float16* ap = &lA[(wave * 16 + fRow) * 32 + kb];
        fa.u[0] = *(const uint4*)(ap);        // K = kb..kb+7
        fa.u[1] = *(const uint4*)(ap + 16);   // K = kb+16..kb+23

        #pragma unroll
        for (int nt = 0; nt < NT; ++nt) {
            Frag16 fb;
            const _Float16* bp = &lB[(nt * 16 + fRow) * 32 + kb];
            fb.u[0] = *(const uint4*)(bp);
            fb.u[1] = *(const uint4*)(bp + 16);
            acc[nt] = __builtin_amdgcn_wmma_f32_16x16x32_f16(
                false, fa.h, false, fb.h, (short)0, acc[nt], false, false);
        }
    }

    // epilogue: C/D layout -> VGPR i is row i (lanes 0-15) / i+8 (lanes 16-31)
    const int colL = lane & 15;
    const int rHi  = (lane >> 4) * 8;
    #pragma unroll
    for (int nt = 0; nt < NT; ++nt) {
        int col = nt * 16 + colL;
        float bv = bias[col];
        #pragma unroll
        for (int i = 0; i < 8; ++i) {
            int r = rowBase + wave * 16 + rHi + i;
            if (r < nRows) {
                float v = acc[nt][i] + bv;
                if (doRelu) v = fmaxf(v, 0.0f);
                out[(size_t)r * NCOLS + col] = v;
            }
        }
    }
}

// ---------------------------------------------------------------- batchnorm
__global__ void bn_stats(const float* __restrict__ h, float* __restrict__ cs,
                         float* __restrict__ css, int rowsPerBlock, int nRows) {
    int col = threadIdx.x;                     // 256 cols
    int r0 = blockIdx.x * rowsPerBlock;
    int r1 = min(r0 + rowsPerBlock, nRows);
    float s = 0.0f, ss = 0.0f;
    for (int r = r0; r < r1; ++r) {
        float v = h[(size_t)r * HID + col];
        s += v; ss += v * v;
    }
    atomicAdd(&cs[col], s);
    atomicAdd(&css[col], ss);
}

__global__ void bn_finalize(const float* __restrict__ cs, const float* __restrict__ css,
                            const float* __restrict__ g, const float* __restrict__ b,
                            float* __restrict__ sc, float* __restrict__ sh, int n) {
    int c = threadIdx.x;
    float inv = 1.0f / (float)n;
    float mu  = cs[c] * inv;
    float var = css[c] * inv - mu * mu;        // population variance (ddof=0)
    float s   = g[c] * rsqrtf(var + BN_EPS);
    sc[c] = s;
    sh[c] = b[c] - mu * s;
}

__global__ void bn_apply(const float* __restrict__ h, const float* __restrict__ sc,
                         const float* __restrict__ sh, float* __restrict__ out, int n4) {
    int i = blockIdx.x * blockDim.x + threadIdx.x;
    if (i >= n4) return;
    float4 v = ((const float4*)h)[i];
    int c = (i * 4) & (HID - 1);
    float4 r;
    r.x = fmaxf(fmaf(v.x, sc[c + 0], sh[c + 0]), 0.0f);
    r.y = fmaxf(fmaf(v.y, sc[c + 1], sh[c + 1]), 0.0f);
    r.z = fmaxf(fmaf(v.z, sc[c + 2], sh[c + 2]), 0.0f);
    r.w = fmaxf(fmaf(v.w, sc[c + 3], sh[c + 3]), 0.0f);
    ((float4*)out)[i] = r;
}

// ---------------------------------------------------------------- final 128->2
__global__ __launch_bounds__(128) void final_mlp(
    const float* __restrict__ h4, const float* __restrict__ w,
    const float* __restrict__ b2, float* __restrict__ out, int nRows) {
    int n = blockIdx.x;
    int k = threadIdx.x;                       // 0..127
    __shared__ float s0[128], s1[128];
    float v = h4[(size_t)n * 128 + k];
    s0[k] = v * w[k * 2 + 0];
    s1[k] = v * w[k * 2 + 1];
    __syncthreads();
    for (int off = 64; off > 0; off >>= 1) {
        if (k < off) { s0[k] += s0[k + off]; s1[k] += s1[k + off]; }
        __syncthreads();
    }
    if (k == 0) {
        out[(size_t)n * 2 + 0] = s0[0] + b2[0];
        out[(size_t)n * 2 + 1] = s1[0] + b2[1];
    }
}

// ---------------------------------------------------------------- launch
extern "C" void kernel_launch(void* const* d_in, const int* in_sizes, int n_in,
                              void* d_out, int out_size, void* d_ws, size_t ws_size,
                              hipStream_t stream) {
    (void)in_sizes; (void)n_in; (void)out_size; (void)ws_size;

    const float* x   = (const float*)d_in[0];
    const int*   ei  = (const int*)d_in[1];
    const int*   src = ei;
    const int*   dst = ei + N_EDGES;
    const float* wl[3] = {(const float*)d_in[2],  (const float*)d_in[7],  (const float*)d_in[12]};
    const float* wr[3] = {(const float*)d_in[3],  (const float*)d_in[8],  (const float*)d_in[13]};
    const float* bb[3] = {(const float*)d_in[4],  (const float*)d_in[9],  (const float*)d_in[14]};
    const float* gg[3] = {(const float*)d_in[5],  (const float*)d_in[10], (const float*)d_in[15]};
    const float* be[3] = {(const float*)d_in[6],  (const float*)d_in[11], (const float*)d_in[16]};
    const float* cw1 = (const float*)d_in[17];
    const float* cb1 = (const float*)d_in[18];
    const float* cw2 = (const float*)d_in[19];
    const float* cb2 = (const float*)d_in[20];
    float* out = (float*)d_out;

    // workspace layout (floats)
    float* agg  = (float*)d_ws;                          // N*256
    float* bufP = agg  + (size_t)N_NODES * HID;          // N*256 (pre-BN)
    float* bufA = bufP + (size_t)N_NODES * HID;          // N*256 (layer activ.)
    float* deg  = bufA + (size_t)N_NODES * HID;          // N
    float* cs   = deg + N_NODES;                         // 256
    float* css  = cs + HID;                              // 256
    float* sc   = css + HID;                             // 256
    float* sh   = sc + HID;                              // 256

    // degrees (identical for every layer)
    zero_f32<<<(N_NODES + 255) / 256, 256, 0, stream>>>(deg, N_NODES);
    deg_kernel<<<(N_EDGES + 255) / 256, 256, 0, stream>>>(dst, deg, N_EDGES);

    const float* xin = x;
    int D = IN_DIM;
    for (int l = 0; l < 3; ++l) {
        int nAgg = N_NODES * D;
        zero_f32<<<(nAgg + 255) / 256, 256, 0, stream>>>(agg, nAgg);
        zero_f32<<<(2 * HID + 255) / 256, 256, 0, stream>>>(cs, 2 * HID);

        int D4 = D / 4;
        int total = N_EDGES * D4;
        scatter_add<<<(total + 255) / 256, 256, 0, stream>>>(xin, src, dst, agg, D4, total);
        int nv = N_NODES * D4;
        norm_agg<<<(nv + 255) / 256, 256, 0, stream>>>(agg, deg, D4, nv);

        sage_gemm<HID><<<(N_NODES + 63) / 64, 128, 0, stream>>>(
            agg, xin, wl[l], wr[l], bb[l], bufP, N_NODES, D, D, 0);

        bn_stats<<<250, 256, 0, stream>>>(bufP, cs, css, 400, N_NODES);
        bn_finalize<<<1, 256, 0, stream>>>(cs, css, gg[l], be[l], sc, sh, N_NODES);
        int n4 = N_NODES * HID / 4;
        bn_apply<<<(n4 + 255) / 256, 256, 0, stream>>>(bufP, sc, sh, bufA, n4);

        xin = bufA;
        D = HID;
    }

    // classifier: 256 -> 128 (WMMA, +bias, ReLU), then 128 -> 2
    sage_gemm<128><<<(N_NODES + 63) / 64, 128, 0, stream>>>(
        bufA, nullptr, cw1, nullptr, cb1, bufP, N_NODES, HID, 0, 1);
    final_mlp<<<N_NODES, 128, 0, stream>>>(bufP, cw2, cb2, out, N_NODES);
}
